// ConditionalVAE_31456340476188
// MI455X (gfx1250) — matrix operand order repaired and checked by reading, not compiled, not run on hardware
//
#include <hip/hip_runtime.h>

// ---------------------------------------------------------------------------
// ConditionalVAE forward for MI455X (gfx1250): bf16 WMMA GEMMs, f32 epilogues.
// Compute-bound (~438 GFLOP vs ~0.3 GB traffic) -> v_wmma_f32_16x16x32_bf16.
// Weights pre-packed once into B-fragment panel layout; hot loop is pure
// b128 global->LDS->fragment traffic + WMMA, double-buffered.
// ---------------------------------------------------------------------------

#define BATCH 8192
#define DIN   512
#define DCOND 256
#define DHID  1024
#define DLAT  128
#define NEXP  4

typedef __attribute__((ext_vector_type(16))) __bf16 v16bf;
typedef __attribute__((ext_vector_type(8)))  __bf16 v8bf;
typedef __attribute__((ext_vector_type(8)))  float  v8f;

#define CAT16(lo, hi) __builtin_shufflevector(lo, hi, 0,1,2,3,4,5,6,7,8,9,10,11,12,13,14,15)

__device__ __forceinline__ float leakyf(float x) { return x > 0.f ? x : 0.01f * x; }
__device__ __forceinline__ float clip5(float x)  { return fminf(5.f, fmaxf(-5.f, x)); }

// ------------------------- weight pack: f32 [K][N] -> panel uint [n][k2] ---
// Wt[n*(K/2)+k2] = bf16(W[2*k2][n]) | bf16(W[2*k2+1][n])<<16
// (dword sequence along k2 == B-fragment element order; tiles load as b128)
__global__ void pack_weight_kernel(const float* __restrict__ W,
                                   unsigned int* __restrict__ Wt, int K, int N) {
  int K2 = K >> 1;
  int nchunks = K2 >> 3;  // 8 k2-dwords per thread
  int t = blockIdx.x * blockDim.x + threadIdx.x;
  if (t >= N * nchunks) return;
  int n = t % N;           // n fastest -> coalesced f32 reads
  int k2base = (t / N) * 8;
  unsigned int outv[8];
#pragma unroll
  for (int j = 0; j < 8; j++) {
    int k2 = k2base + j;
    __bf16 lo = (__bf16)W[(long)(2 * k2) * N + n];
    __bf16 hi = (__bf16)W[(long)(2 * k2 + 1) * N + n];
    unsigned short ulo = __builtin_bit_cast(unsigned short, lo);
    unsigned short uhi = __builtin_bit_cast(unsigned short, hi);
    outv[j] = (unsigned int)ulo | ((unsigned int)uhi << 16);
  }
  *(uint4*)&Wt[(long)n * K2 + k2base]     = *(const uint4*)&outv[0];
  *(uint4*)&Wt[(long)n * K2 + k2base + 4] = *(const uint4*)&outv[4];
}

// ------------------------- input normalization + concat layout -------------
__global__ void prep_kernel(const float* __restrict__ x, const float* __restrict__ w,
                            const float* __restrict__ im, const float* __restrict__ iv,
                            const float* __restrict__ cm, const float* __restrict__ cv,
                            __bf16* __restrict__ A0,
                            __bf16* __restrict__ buf1, __bf16* __restrict__ buf2,
                            __bf16* __restrict__ decA, __bf16* __restrict__ decB,
                            float* __restrict__ wnf) {
  int idx = blockIdx.x * blockDim.x + threadIdx.x;
  if (idx >= BATCH * (DIN + DCOND)) return;
  int row = idx / (DIN + DCOND);
  int col = idx % (DIN + DCOND);
  if (col < DIN) {
    float v = (x[row * DIN + col] - im[col]) * rsqrtf(iv[col] + 1e-4f);
    v = clip5(v);
    A0[(long)row * (DIN + DCOND) + col] = (__bf16)v;
  } else {
    int c = col - DIN;
    float v = (w[row * DCOND + c] - cm[c]) * rsqrtf(cv[c] + 1e-4f);
    v = clip5(v);
    if (v != v) v = 0.f;  // NaN -> 0 per reference
    __bf16 bv = (__bf16)v;
    A0[(long)row * (DIN + DCOND) + col] = bv;
    buf1[(long)row * 1280 + c] = bv;
    buf2[(long)row * 1280 + c] = bv;
    decA[(long)row * 1408 + DLAT + c] = bv;
    decB[(long)row * 1408 + DLAT + c] = bv;
    wnf[row * DCOND + c] = v;
  }
}

// ------------------------- WMMA GEMM (single weight) -----------------------
// C[M,N] = A[M,K] @ W[K,N] + bias ; ACT: 0 none, 1 leaky, 2 clip(+-5).
// Tile 128x128, BK=32, 8 waves, wave tile 32x64 (2x4 WMMA f32_16x16x32_bf16).
// Double-buffered LDS: prefetch tile s+1 to regs during compute of tile s.
template <int ACT, bool OUTB, bool OUTF>
__global__ __launch_bounds__(256)
void gemm_bf16_wmma(const __bf16* __restrict__ A, int lda, int K,
                    const unsigned int* __restrict__ Wt,  // packed [n][k2]
                    const float* __restrict__ bias,
                    __bf16* __restrict__ outb, int ldob, int ocol,
                    float* __restrict__ outf, int ldof) {
  __shared__ __bf16       As[2][128 * 40];  // [row][k], 80B pitch
  __shared__ unsigned int Ws[2][128 * 20];  // [n][k2] packed

  const int tid  = threadIdx.x;
  const int lane = tid & 31;
  const int wid  = tid >> 5;
  const int wm   = wid & 3;
  const int wn   = wid >> 2;
  const long rowBase = (long)blockIdx.x * 128;
  const int  colBase = blockIdx.y * 128;
  const int K2 = K >> 1;
  const int arow = tid >> 1, aseg = (tid & 1) * 16;
  const int wcol = tid >> 1, wk = (tid & 1) * 8;  // 8 k2-dwords per thread
  const int lh = lane >> 4, ll = lane & 15;

  const v8f vzero = {0.f,0.f,0.f,0.f,0.f,0.f,0.f,0.f};
  v8f acc[2][4];
#pragma unroll
  for (int i = 0; i < 2; i++)
#pragma unroll
    for (int j = 0; j < 4; j++) acc[i][j] = vzero;

  uint4 ra0, ra1, rw0, rw1;
  auto prefetch = [&](int k0) {
    const __bf16* asrc = A + (rowBase + arow) * lda + k0 + aseg;
    ra0 = *(const uint4*)asrc;
    ra1 = *(const uint4*)(asrc + 8);
    const unsigned int* wsrc = Wt + (long)(colBase + wcol) * K2 + (k0 >> 1) + wk;
    rw0 = *(const uint4*)wsrc;
    rw1 = *(const uint4*)(wsrc + 4);
  };
  auto stage = [&](int buf) {
    *(uint4*)&As[buf][arow * 40 + aseg]     = ra0;
    *(uint4*)&As[buf][arow * 40 + aseg + 8] = ra1;
    *(uint4*)&Ws[buf][wcol * 20 + wk]       = rw0;
    *(uint4*)&Ws[buf][wcol * 20 + wk + 4]   = rw1;
  };

  const int ksteps = K / 32;
  prefetch(0);
  stage(0);
  for (int s = 0; s < ksteps; s++) {
    __syncthreads();
    if (s + 1 < ksteps) prefetch((s + 1) * 32);
    const __bf16*       Ab = As[s & 1];
    const unsigned int* Wb = Ws[s & 1];
    v16bf afrag[2];
#pragma unroll
    for (int tm = 0; tm < 2; tm++) {
      int m = wm * 32 + tm * 16 + ll;
      v8bf lo = *(const v8bf*)&Ab[m * 40 + lh * 8];
      v8bf hi = *(const v8bf*)&Ab[m * 40 + 16 + lh * 8];
      afrag[tm] = CAT16(lo, hi);
    }
#pragma unroll
    for (int tn = 0; tn < 4; tn++) {
      int n = wn * 64 + tn * 16 + ll;
      v8bf lo = *(const v8bf*)&Wb[n * 20 + lh * 4];
      v8bf hi = *(const v8bf*)&Wb[n * 20 + 8 + lh * 4];
      v16bf bfrag = CAT16(lo, hi);
#pragma unroll
      for (int tm = 0; tm < 2; tm++)
        acc[tm][tn] = __builtin_amdgcn_wmma_f32_16x16x32_bf16(
            false, afrag[tm], false, bfrag, (short)0, acc[tm][tn], false, false);
    }
    if (s + 1 < ksteps) stage((s + 1) & 1);
  }

  // epilogue: C/D layout -> row = wm*32+tm*16+lh*8+v, col = wn*64+tn*16+ll
#pragma unroll
  for (int tm = 0; tm < 2; tm++)
#pragma unroll
    for (int tn = 0; tn < 4; tn++) {
      int n = colBase + wn * 64 + tn * 16 + ll;
      float bv = bias[n];
#pragma unroll
      for (int v = 0; v < 8; v++) {
        long r = rowBase + wm * 32 + tm * 16 + lh * 8 + v;
        float val = acc[tm][tn][v] + bv;
        if constexpr (ACT == 1) val = leakyf(val);
        if constexpr (ACT == 2) val = clip5(val);
        if constexpr (OUTB) outb[r * ldob + ocol + n] = (__bf16)val;
        if constexpr (OUTF) outf[r * ldof + n] = val;
      }
    }
}

// ------------------------- WMMA soft-MoE GEMM ------------------------------
// out = sum_e coeff[b,e] * (A @ W_e + b_e); flattened (e,k) software pipeline.
template <int ACT, bool OUTB, bool OUTF>
__global__ __launch_bounds__(256)
void moe_gemm_bf16_wmma(const __bf16* __restrict__ A, int lda, int K,
                        const unsigned int* __restrict__ Wt, long wstride,  // uints/expert
                        const float* __restrict__ bias, int bstride,
                        const float* __restrict__ coeff,
                        __bf16* __restrict__ outb, int ldob, int ocol,
                        float* __restrict__ outf, int ldof) {
  __shared__ __bf16       As[2][128 * 40];
  __shared__ unsigned int Ws[2][128 * 20];
  __shared__ float        Cs[128 * 4];

  const int tid  = threadIdx.x;
  const int lane = tid & 31;
  const int wid  = tid >> 5;
  const int wm   = wid & 3;
  const int wn   = wid >> 2;
  const long rowBase = (long)blockIdx.x * 128;
  const int  colBase = blockIdx.y * 128;
  const int K2 = K >> 1;
  const int arow = tid >> 1, aseg = (tid & 1) * 16;
  const int wcol = tid >> 1, wk = (tid & 1) * 8;
  const int lh = lane >> 4, ll = lane & 15;

  if (tid < 128) *(float4*)&Cs[tid * 4] = *(const float4*)&coeff[(rowBase + tid) * 4];

  const v8f vzero = {0.f,0.f,0.f,0.f,0.f,0.f,0.f,0.f};
  v8f tot[2][4], acc[2][4];
#pragma unroll
  for (int i = 0; i < 2; i++)
#pragma unroll
    for (int j = 0; j < 4; j++) { tot[i][j] = vzero; acc[i][j] = vzero; }

  const int ksteps = K / 32;
  const int total  = NEXP * ksteps;

  uint4 ra0, ra1, rw0, rw1;
  auto prefetch = [&](int step) {
    int e  = step / ksteps;
    int k0 = (step - e * ksteps) * 32;
    const __bf16* asrc = A + (rowBase + arow) * lda + k0 + aseg;
    ra0 = *(const uint4*)asrc;
    ra1 = *(const uint4*)(asrc + 8);
    const unsigned int* wsrc =
        Wt + (long)e * wstride + (long)(colBase + wcol) * K2 + (k0 >> 1) + wk;
    rw0 = *(const uint4*)wsrc;
    rw1 = *(const uint4*)(wsrc + 4);
  };
  auto stage = [&](int buf) {
    *(uint4*)&As[buf][arow * 40 + aseg]     = ra0;
    *(uint4*)&As[buf][arow * 40 + aseg + 8] = ra1;
    *(uint4*)&Ws[buf][wcol * 20 + wk]       = rw0;
    *(uint4*)&Ws[buf][wcol * 20 + wk + 4]   = rw1;
  };

  prefetch(0);
  stage(0);
  for (int s = 0; s < total; s++) {
    __syncthreads();
    if (s + 1 < total) prefetch(s + 1);
    const __bf16*       Ab = As[s & 1];
    const unsigned int* Wb = Ws[s & 1];
    v16bf afrag[2];
#pragma unroll
    for (int tm = 0; tm < 2; tm++) {
      int m = wm * 32 + tm * 16 + ll;
      v8bf lo = *(const v8bf*)&Ab[m * 40 + lh * 8];
      v8bf hi = *(const v8bf*)&Ab[m * 40 + 16 + lh * 8];
      afrag[tm] = CAT16(lo, hi);
    }
#pragma unroll
    for (int tn = 0; tn < 4; tn++) {
      int n = wn * 64 + tn * 16 + ll;
      v8bf lo = *(const v8bf*)&Wb[n * 20 + lh * 4];
      v8bf hi = *(const v8bf*)&Wb[n * 20 + 8 + lh * 4];
      v16bf bfrag = CAT16(lo, hi);
#pragma unroll
      for (int tm = 0; tm < 2; tm++)
        acc[tm][tn] = __builtin_amdgcn_wmma_f32_16x16x32_bf16(
            false, afrag[tm], false, bfrag, (short)0, acc[tm][tn], false, false);
    }
    // expert boundary: fold bias, scale by coeff[row,e], accumulate, reset
    if (((s + 1) % ksteps) == 0) {
      int e = s / ksteps;
      const float* be = bias + (long)e * bstride;
#pragma unroll
      for (int tm = 0; tm < 2; tm++)
#pragma unroll
        for (int tn = 0; tn < 4; tn++) {
          int n = colBase + wn * 64 + tn * 16 + ll;
          float bv = be[n];
#pragma unroll
          for (int v = 0; v < 8; v++) {
            int rl = wm * 32 + tm * 16 + lh * 8 + v;
            tot[tm][tn][v] += Cs[rl * 4 + e] * (acc[tm][tn][v] + bv);
          }
          acc[tm][tn] = vzero;
        }
    }
    if (s + 1 < total) stage((s + 1) & 1);
  }

#pragma unroll
  for (int tm = 0; tm < 2; tm++)
#pragma unroll
    for (int tn = 0; tn < 4; tn++) {
      int n = colBase + wn * 64 + tn * 16 + ll;
#pragma unroll
      for (int v = 0; v < 8; v++) {
        long r = rowBase + wm * 32 + tm * 16 + lh * 8 + v;
        float val = tot[tm][tn][v];
        if constexpr (ACT == 1) val = leakyf(val);
        if constexpr (OUTB) outb[r * ldob + ocol + n] = (__bf16)val;
        if constexpr (OUTF) outf[r * ldof + n] = val;
      }
    }
}

// ------------------------- reparameterize ----------------------------------
__global__ void reparam_kernel(const float* __restrict__ mu, const float* __restrict__ lv,
                               const float* __restrict__ eps, float* __restrict__ z,
                               __bf16* __restrict__ decA, __bf16* __restrict__ decB) {
  int i = blockIdx.x * blockDim.x + threadIdx.x;
  if (i >= BATCH * DLAT) return;
  int row = i / DLAT, c = i % DLAT;
  float zz = mu[i] + eps[i] * expf(0.5f * lv[i]);
  z[i] = zz;
  __bf16 zb = (__bf16)zz;
  decA[(long)row * 1408 + c] = zb;
  decB[(long)row * 1408 + c] = zb;
}

// ------------------------- tiny gate network (per-row) ---------------------
__global__ __launch_bounds__(64)
void gate_kernel(const float* __restrict__ z, const float* __restrict__ wnf,
                 const float* __restrict__ w0, const float* __restrict__ b0,
                 const float* __restrict__ w1, const float* __restrict__ b1,
                 const float* __restrict__ w2, const float* __restrict__ b2,
                 float* __restrict__ coeff) {
  __shared__ float h0[64], h1[64], lg[4];
  int row = blockIdx.x, t = threadIdx.x;
  const float* zr = z + (long)row * DLAT;
  const float* cr = wnf + (long)row * DCOND;
  float acc = b0[t];
  for (int i = 0; i < DLAT; i++)  acc += zr[i] * w0[i * 64 + t];
  for (int i = 0; i < DCOND; i++) acc += cr[i] * w0[(DLAT + i) * 64 + t];
  h0[t] = leakyf(acc);
  __syncthreads();
  acc = b1[t];
  for (int i = 0; i < 64; i++) acc += h0[i] * w1[i * 64 + t];
  h1[t] = leakyf(acc);
  __syncthreads();
  if (t < 4) {
    float a2 = b2[t];
    for (int i = 0; i < 64; i++) a2 += h1[i] * w2[i * 4 + t];
    lg[t] = a2;
  }
  __syncthreads();
  if (t == 0) {
    float m = fmaxf(fmaxf(lg[0], lg[1]), fmaxf(lg[2], lg[3]));
    float e0 = expf(lg[0] - m), e1 = expf(lg[1] - m);
    float e2 = expf(lg[2] - m), e3 = expf(lg[3] - m);
    float s = e0 + e1 + e2 + e3;
    coeff[row * 4 + 0] = e0 / s; coeff[row * 4 + 1] = e1 / s;
    coeff[row * 4 + 2] = e2 / s; coeff[row * 4 + 3] = e3 / s;
  }
}

// ---------------------------------------------------------------------------
extern "C" void kernel_launch(void* const* d_in, const int* in_sizes, int n_in,
                              void* d_out, int out_size, void* d_ws, size_t ws_size,
                              hipStream_t stream) {
  const float* x      = (const float*)d_in[0];
  const float* w      = (const float*)d_in[1];
  const float* eps    = (const float*)d_in[2];
  const float* rim    = (const float*)d_in[3];
  const float* riv    = (const float*)d_in[4];
  const float* rcm    = (const float*)d_in[5];
  const float* rcv    = (const float*)d_in[6];
  const float* enc_w0 = (const float*)d_in[7];
  const float* enc_b0 = (const float*)d_in[8];
  const float* enc_w1 = (const float*)d_in[9];
  const float* enc_b1 = (const float*)d_in[10];
  const float* enc_w2 = (const float*)d_in[11];
  const float* enc_b2 = (const float*)d_in[12];
  const float* enc_w3 = (const float*)d_in[13];
  const float* enc_b3 = (const float*)d_in[14];
  const float* mu_w   = (const float*)d_in[15];
  const float* mu_b   = (const float*)d_in[16];
  const float* lv_w   = (const float*)d_in[17];
  const float* lv_b   = (const float*)d_in[18];
  const float* g_w0   = (const float*)d_in[19];
  const float* g_b0   = (const float*)d_in[20];
  const float* g_w1   = (const float*)d_in[21];
  const float* g_b1   = (const float*)d_in[22];
  const float* g_w2   = (const float*)d_in[23];
  const float* g_b2   = (const float*)d_in[24];
  const float* dec_w0 = (const float*)d_in[25];
  const float* dec_b0 = (const float*)d_in[26];
  const float* dec_w1 = (const float*)d_in[27];
  const float* dec_b1 = (const float*)d_in[28];
  const float* dec_w2 = (const float*)d_in[29];
  const float* dec_b2 = (const float*)d_in[30];
  const float* dec_w3 = (const float*)d_in[31];
  const float* dec_b3 = (const float*)d_in[32];
  const float* dec_w4 = (const float*)d_in[33];
  const float* dec_b4 = (const float*)d_in[34];

  float* out = (float*)d_out;
  const long Z_OFF   = 0;
  const long OUT_OFF = (long)BATCH * DLAT;
  const long MU_OFF  = OUT_OFF + (long)BATCH * DIN;
  const long LV_OFF  = MU_OFF + (long)BATCH * DLAT;

  char* ws = (char*)d_ws;
  size_t off = 0;
  auto alloc = [&](size_t bytes) -> char* {
    char* p = ws + off;
    off += (bytes + 255) & ~(size_t)255;
    return p;
  };
  // packed weights (uint per 2 K-values; same bytes as bf16 K*N)
  unsigned int* w0p  = (unsigned int*)alloc((size_t)768  * 1024 * 2);
  unsigned int* w1p  = (unsigned int*)alloc((size_t)1280 * 1024 * 2);
  unsigned int* w2p  = (unsigned int*)alloc((size_t)1280 * 1024 * 2);
  unsigned int* w3p  = (unsigned int*)alloc((size_t)1280 * 1024 * 2);
  unsigned int* mup  = (unsigned int*)alloc((size_t)1024 * 128 * 2);
  unsigned int* lvp  = (unsigned int*)alloc((size_t)1024 * 128 * 2);
  unsigned int* d0p  = (unsigned int*)alloc((size_t)4 * 384  * 1024 * 2);
  unsigned int* d1p  = (unsigned int*)alloc((size_t)4 * 1408 * 1024 * 2);
  unsigned int* d2p  = (unsigned int*)alloc((size_t)4 * 1408 * 1024 * 2);
  unsigned int* d3p  = (unsigned int*)alloc((size_t)4 * 1408 * 1024 * 2);
  unsigned int* d4p  = (unsigned int*)alloc((size_t)4 * 1408 * 512  * 2);
  __bf16* A0     = (__bf16*)alloc((size_t)BATCH * 768  * 2);
  __bf16* buf1   = (__bf16*)alloc((size_t)BATCH * 1280 * 2);
  __bf16* buf2   = (__bf16*)alloc((size_t)BATCH * 1280 * 2);
  __bf16* hleaky = (__bf16*)alloc((size_t)BATCH * 1024 * 2);
  __bf16* decA   = (__bf16*)alloc((size_t)BATCH * 1408 * 2);
  __bf16* decB   = (__bf16*)alloc((size_t)BATCH * 1408 * 2);
  float*  wnf    = (float*)alloc((size_t)BATCH * DCOND * 4);
  float*  coeff  = (float*)alloc((size_t)BATCH * 4 * 4);

  auto pack = [&](const float* src, unsigned int* dst, int K, int N) {
    int nthreads = N * ((K / 2) / 8);
    pack_weight_kernel<<<(nthreads + 255) / 256, 256, 0, stream>>>(src, dst, K, N);
  };
  pack(enc_w0, w0p, 768, 1024);
  pack(enc_w1, w1p, 1280, 1024);
  pack(enc_w2, w2p, 1280, 1024);
  pack(enc_w3, w3p, 1280, 1024);
  pack(mu_w, mup, 1024, 128);
  pack(lv_w, lvp, 1024, 128);
  for (int e = 0; e < NEXP; e++) {
    pack(dec_w0 + (long)e * 384 * 1024,  d0p + (long)e * 192 * 1024, 384, 1024);
    pack(dec_w1 + (long)e * 1408 * 1024, d1p + (long)e * 704 * 1024, 1408, 1024);
    pack(dec_w2 + (long)e * 1408 * 1024, d2p + (long)e * 704 * 1024, 1408, 1024);
    pack(dec_w3 + (long)e * 1408 * 1024, d3p + (long)e * 704 * 1024, 1408, 1024);
    pack(dec_w4 + (long)e * 1408 * 512,  d4p + (long)e * 704 * 512,  1408, 512);
  }

  prep_kernel<<<(BATCH * (DIN + DCOND) + 255) / 256, 256, 0, stream>>>(
      x, w, rim, riv, rcm, rcv, A0, buf1, buf2, decA, decB, wnf);

  dim3 blk(256);
  // encoder (linear-only chain, concat via prefilled wn columns)
  gemm_bf16_wmma<0, true, false><<<dim3(BATCH / 128, 1024 / 128), blk, 0, stream>>>(
      A0, 768, 768, w0p, enc_b0, buf1, 1280, 256, nullptr, 0);
  gemm_bf16_wmma<0, true, false><<<dim3(BATCH / 128, 1024 / 128), blk, 0, stream>>>(
      buf1, 1280, 1280, w1p, enc_b1, buf2, 1280, 256, nullptr, 0);
  gemm_bf16_wmma<0, true, false><<<dim3(BATCH / 128, 1024 / 128), blk, 0, stream>>>(
      buf2, 1280, 1280, w2p, enc_b2, buf1, 1280, 256, nullptr, 0);
  // enc3: store leaky(h) (only consumer is mu/lv heads)
  gemm_bf16_wmma<1, true, false><<<dim3(BATCH / 128, 1024 / 128), blk, 0, stream>>>(
      buf1, 1280, 1280, w3p, enc_b3, hleaky, 1024, 0, nullptr, 0);
  // heads
  gemm_bf16_wmma<0, false, true><<<dim3(BATCH / 128, 1), blk, 0, stream>>>(
      hleaky, 1024, 1024, mup, mu_b, nullptr, 0, 0, out + MU_OFF, 128);
  gemm_bf16_wmma<2, false, true><<<dim3(BATCH / 128, 1), blk, 0, stream>>>(
      hleaky, 1024, 1024, lvp, lv_b, nullptr, 0, 0, out + LV_OFF, 128);
  // z = mu + eps*exp(0.5*logvar); seed z columns of decoder buffers
  reparam_kernel<<<(BATCH * DLAT + 255) / 256, 256, 0, stream>>>(
      out + MU_OFF, out + LV_OFF, eps, out + Z_OFF, decA, decB);
  // gate -> softmax coefficients
  gate_kernel<<<BATCH, 64, 0, stream>>>(out + Z_OFF, wnf, g_w0, g_b0, g_w1, g_b1,
                                        g_w2, g_b2, coeff);
  // soft-MoE decoder: layer0 reads [z,c]=first 384 cols; layers1-4 read 1408
  moe_gemm_bf16_wmma<1, true, false><<<dim3(BATCH / 128, 1024 / 128), blk, 0, stream>>>(
      decA, 1408, 384, d0p, 192L * 1024, dec_b0, 1024, coeff,
      decB, 1408, 384, nullptr, 0);
  moe_gemm_bf16_wmma<1, true, false><<<dim3(BATCH / 128, 1024 / 128), blk, 0, stream>>>(
      decB, 1408, 1408, d1p, 704L * 1024, dec_b1, 1024, coeff,
      decA, 1408, 384, nullptr, 0);
  moe_gemm_bf16_wmma<1, true, false><<<dim3(BATCH / 128, 1024 / 128), blk, 0, stream>>>(
      decA, 1408, 1408, d2p, 704L * 1024, dec_b2, 1024, coeff,
      decB, 1408, 384, nullptr, 0);
  moe_gemm_bf16_wmma<1, true, false><<<dim3(BATCH / 128, 1024 / 128), blk, 0, stream>>>(
      decB, 1408, 1408, d3p, 704L * 1024, dec_b3, 1024, coeff,
      decA, 1408, 384, nullptr, 0);
  moe_gemm_bf16_wmma<0, false, true><<<dim3(BATCH / 128, 512 / 128), blk, 0, stream>>>(
      decA, 1408, 1408, d4p, 704L * 512, dec_b4, 512, coeff,
      nullptr, 0, 0, out + OUT_OFF, 512);
}
